// RelationChainDecoder_42949672960267
// MI455X (gfx1250) — matrix operand order repaired and checked by reading, not compiled, not run on hardware
//
#include <hip/hip_runtime.h>
#include <hip/hip_bf16.h>

// ---------------------------------------------------------------------------
// RelationChainDecoder on gfx1250 (MI455X).
// GEMMs on the matrix pipe: v_wmma_f32_16x16x32_bf16, f32 accumulate,
// bf16 operands packed via v_perm_b32 into double-buffered LDS tiles;
// 8 WMMAs per wave per K-step, global loads pipelined one step ahead.
// ---------------------------------------------------------------------------

typedef __attribute__((ext_vector_type(16))) __bf16 v16bf;
typedef __attribute__((ext_vector_type(8)))  float  v8f;

#define B_SZ    2048
#define HID     1024
#define EMB     1024
#define MAX_LEN 64
#define VOCAB   20000

// f32x2 -> packed bf16x2. Prefer HW packed convert when declared; otherwise
// round-half-up (+0x8000) and pack both high halves with one v_perm_b32.
#if __has_builtin(__builtin_amdgcn_cvt_pk_bf16_f32)
__device__ __forceinline__ unsigned int pack2bf(float lo, float hi) {
    auto p = __builtin_amdgcn_cvt_pk_bf16_f32(lo, hi);
    return __builtin_bit_cast(unsigned int, p);
}
#else
__device__ __forceinline__ unsigned int pack2bf(float lo, float hi) {
    unsigned int ul = __float_as_uint(lo) + 0x8000u;
    unsigned int uh = __float_as_uint(hi) + 0x8000u;
    // result bytes {uh[3],uh[2],ul[3],ul[2]}  ->  {bf16(hi), bf16(lo)}
    return __builtin_amdgcn_perm(uh, ul, 0x07060302u);
}
#endif

union ABFrag { v16bf v; unsigned int u[8]; };

// Per-thread register staging for one K-step tile (A: 32 f32, W: 8 f32).
struct Stage { float4 a[8]; float4 w[2]; };

__device__ __forceinline__ void load_stage(Stage& s,
        const float* __restrict__ A, const float* __restrict__ W,
        int K, int N, int blockM, int blockN, int tid, int k0) {
    // A tile: 256 rows x 32 k, one row (128B contiguous) per thread.
    const float4* asrc = (const float4*)(A + (long long)(blockM + tid) * K + k0);
    #pragma unroll
    for (int j = 0; j < 8; ++j) s.a[j] = asrc[j];
    // W tile: 64 rows x 32 k, 4 threads per row (8 floats each), N-guarded.
    const int roww = tid >> 2, kk = (tid & 3) << 3;
    const int gn = blockN + roww;
    if (gn < N) {
        const float4* wsrc = (const float4*)(W + (long long)gn * K + k0 + kk);
        s.w[0] = wsrc[0];
        s.w[1] = wsrc[1];
    } else {
        s.w[0] = make_float4(0.f, 0.f, 0.f, 0.f);
        s.w[1] = make_float4(0.f, 0.f, 0.f, 0.f);
    }
}

__device__ __forceinline__ void store_stage(const Stage& s,
        unsigned int* __restrict__ sAbuf, unsigned int* __restrict__ sWbuf,
        int tid) {
    unsigned int* da = sAbuf + tid * 17;
    #pragma unroll
    for (int j = 0; j < 8; ++j) {
        da[2 * j]     = pack2bf(s.a[j].x, s.a[j].y);
        da[2 * j + 1] = pack2bf(s.a[j].z, s.a[j].w);
    }
    unsigned int* dw = sWbuf + (tid >> 2) * 17 + ((tid & 3) << 2);
    dw[0] = pack2bf(s.w[0].x, s.w[0].y);
    dw[1] = pack2bf(s.w[0].z, s.w[0].w);
    dw[2] = pack2bf(s.w[1].x, s.w[1].y);
    dw[3] = pack2bf(s.w[1].z, s.w[1].w);
}

// ---------------------------------------------------------------------------
// C[M,N] = act(A[M,K] @ W[N,K]^T + bias)   (torch x @ w.T convention)
// M = gridDim.y*256 (2048 here), K multiple of 32, N multiple of 16.
// Block: 256 threads = 8 waves. Block tile 256x64; wave tile 32x64
// (2 A-frags x 4 B-frags = 8 WMMAs per K-step). Double-buffered LDS.
// ---------------------------------------------------------------------------
__global__ __launch_bounds__(256)
void gemm_wmma_bf16(const float* __restrict__ A, int K,
                    const float* __restrict__ W, int N,
                    const float* __restrict__ bias,
                    float* __restrict__ C, int act) {
    // Packed bf16 pairs; row pitch 17 uints (=34 bf16) breaks bank strides.
    __shared__ unsigned int sA[2][256 * 17];   // 2 x 17408 B
    __shared__ unsigned int sW[2][64 * 17];    // 2 x  4352 B

    const int tid    = threadIdx.x;
    const int wave   = tid >> 5;
    const int lane   = tid & 31;
    const int r      = lane & 15;
    const int hlf    = lane >> 4;
    const int blockM = blockIdx.y * 256;
    const int blockN = blockIdx.x * 64;

    v8f acc[2][4];
    #pragma unroll
    for (int mt = 0; mt < 2; ++mt)
        #pragma unroll
        for (int nt = 0; nt < 4; ++nt)
            #pragma unroll
            for (int e = 0; e < 8; ++e) acc[mt][nt][e] = 0.0f;

    Stage st;
    load_stage(st, A, W, K, N, blockM, blockN, tid, 0);
    store_stage(st, sA[0], sW[0], tid);
    __syncthreads();

    const int nsteps = K >> 5;
    for (int s = 0; s < nsteps; ++s) {
        const int buf = s & 1;
        // Prefetch next K-step's global data while this step computes.
        if (s + 1 < nsteps)
            load_stage(st, A, W, K, N, blockM, blockN, tid, (s + 1) << 5);

        // A fragments (ISA 16-bit A layout: VGPR0-3 = K[hlf*8..+7],
        // VGPR4-7 = K[16+hlf*8..+7]; lane r = row).
        const unsigned int* pa = &sA[buf][(wave * 32 + r) * 17];
        ABFrag a0, a1;
        #pragma unroll
        for (int i = 0; i < 4; ++i) {
            a0.u[i]     = pa[(hlf << 2) + i];
            a0.u[4 + i] = pa[8 + (hlf << 2) + i];
            a1.u[i]     = pa[16 * 17 + (hlf << 2) + i];
            a1.u[4 + i] = pa[16 * 17 + 8 + (hlf << 2) + i];
        }
        // B fragments (same K layout as A: VGPR0-3 = K[hlf*8..+7], VGPR4-7 = K[16+hlf*8..+7]; lane r = col).
        #pragma unroll
        for (int nt = 0; nt < 4; ++nt) {
            ABFrag bfr;
            const unsigned int* pb = &sW[buf][(nt * 16 + r) * 17 + (hlf << 2)];
            #pragma unroll
            for (int i = 0; i < 8; ++i) bfr.u[i] = pb[i + ((i >> 2) << 2)];
            acc[0][nt] = __builtin_amdgcn_wmma_f32_16x16x32_bf16(
                false, a0.v, false, bfr.v, (short)0, acc[0][nt], false, false);
            acc[1][nt] = __builtin_amdgcn_wmma_f32_16x16x32_bf16(
                false, a1.v, false, bfr.v, (short)0, acc[1][nt], false, false);
        }

        if (s + 1 < nsteps) {
            store_stage(st, sA[buf ^ 1], sW[buf ^ 1], tid);
            __syncthreads();
        }
    }

    // Epilogue: C/D layout: element i -> row hlf*8+i, col = r.
    #pragma unroll
    for (int mt = 0; mt < 2; ++mt) {
        #pragma unroll
        for (int nt = 0; nt < 4; ++nt) {
            int col = blockN + nt * 16 + r;
            if (col < N) {
                float bv = bias[col];
                #pragma unroll
                for (int i = 0; i < 8; ++i) {
                    int row = blockM + wave * 32 + mt * 16 + hlf * 8 + i;
                    float v = acc[mt][nt][i] + bv;
                    if (act == 1) v = fmaxf(v, 0.0f);
                    C[(long long)row * N + col] = v;
                }
            }
        }
    }
}

// ---------------------------------------------------------------------------
// Front end: embedding gather, attention logits + softmax, weighted encoder
// reduction, and x_cat = [attn_applied | embedded].  One block per batch row.
// ---------------------------------------------------------------------------
__global__ __launch_bounds__(256)
void attn_front(const int* __restrict__ rel,
                const float* __restrict__ hidden,      // [2,B,H]
                const float* __restrict__ enc,         // [L,B,H] flat
                const float* __restrict__ emb,         // [V+1,E]
                const float* __restrict__ attn_w,      // [L,2H]
                const float* __restrict__ attn_b,      // [L]
                float* __restrict__ x_cat) {           // [B,2H]
    __shared__ float s_in[2 * HID];
    __shared__ float s_part[256];
    __shared__ float s_w[MAX_LEN];

    const int b = blockIdx.x;
    const int t = threadIdx.x;

    const float* erow = emb + (long long)rel[b] * EMB;
    const float* hrow = hidden + (long long)(B_SZ + b) * HID;   // hidden[-1]
    for (int i = t; i < HID; i += 256) {
        s_in[i]       = erow[i];
        s_in[HID + i] = hrow[i];
    }
    __syncthreads();

    // logits: l = t&63, seg = t>>6 covers 512 elems of the 2H dot
    {
        const int l = t & 63, seg = t >> 6;
        const float* wrow = attn_w + l * (2 * HID) + seg * 512;
        const float* xin  = s_in + seg * 512;
        float p = 0.0f;
        #pragma unroll 8
        for (int k = 0; k < 512; ++k) p += wrow[k] * xin[k];
        s_part[t] = p;
    }
    __syncthreads();
    if (t < 64)
        s_w[t] = s_part[t] + s_part[t + 64] + s_part[t + 128] + s_part[t + 192]
               + attn_b[t];
    __syncthreads();
    if (t == 0) {   // 64-wide softmax: serial is negligible
        float m = s_w[0];
        for (int i = 1; i < 64; ++i) m = fmaxf(m, s_w[i]);
        float s = 0.0f;
        for (int i = 0; i < 64; ++i) { float e = __expf(s_w[i] - m); s_w[i] = e; s += e; }
        float inv = 1.0f / s;
        for (int i = 0; i < 64; ++i) s_w[i] *= inv;
    }
    __syncthreads();

    // attn_applied[b,h] = sum_l w[l] * enc_flat[b*L*H + l*H + h]  (coalesced)
    const float* encb = enc + (long long)b * (MAX_LEN * HID);
    float* xrow = x_cat + (long long)b * (2 * HID);
    for (int h = t; h < HID; h += 256) {
        float acc = 0.0f;
        #pragma unroll 8
        for (int l = 0; l < MAX_LEN; ++l) acc += s_w[l] * encb[l * HID + h];
        xrow[h] = acc;
    }
    for (int i = t; i < HID; i += 256) xrow[HID + i] = s_in[i];
}

// ---------------------------------------------------------------------------
// Fused GRU gate combine: h' = (1-z)*n + z*h
// ---------------------------------------------------------------------------
__global__ __launch_bounds__(256)
void gru_gate(const float* __restrict__ gx, const float* __restrict__ gh,
              const float* __restrict__ h, float* __restrict__ h_out) {
    long long i = (long long)blockIdx.x * 256 + threadIdx.x;
    if (i >= (long long)B_SZ * HID) return;
    long long b = i >> 10;
    int       j = (int)(i & 1023);
    const float* gxr = gx + b * 3 * HID;
    const float* ghr = gh + b * 3 * HID;
    float r = 1.0f / (1.0f + __expf(-(gxr[j]           + ghr[j])));
    float z = 1.0f / (1.0f + __expf(-(gxr[HID + j]     + ghr[HID + j])));
    float n = tanhf(gxr[2 * HID + j] + r * ghr[2 * HID + j]);
    float hv = h[i];
    h_out[i] = (1.0f - z) * n + z * hv;
}

// ---------------------------------------------------------------------------
extern "C" void kernel_launch(void* const* d_in, const int* in_sizes, int n_in,
                              void* d_out, int out_size, void* d_ws, size_t ws_size,
                              hipStream_t stream) {
    (void)in_sizes; (void)n_in; (void)out_size; (void)ws_size;

    const int*   rel    = (const int*)  d_in[0];
    const float* hidden = (const float*)d_in[1];
    const float* enc    = (const float*)d_in[2];
    const float* emb    = (const float*)d_in[3];
    const float* attn_w = (const float*)d_in[4];
    const float* attn_b = (const float*)d_in[5];
    const float* comb_w = (const float*)d_in[6];
    const float* comb_b = (const float*)d_in[7];
    const float* hl_w   = (const float*)d_in[8];
    const float* hl_b   = (const float*)d_in[9];
    const float* fc_w   = (const float*)d_in[10];
    const float* fc_b   = (const float*)d_in[11];
    const float* wih0   = (const float*)d_in[12];
    const float* whh0   = (const float*)d_in[13];
    const float* bih0   = (const float*)d_in[14];
    const float* bhh0   = (const float*)d_in[15];
    const float* wih1   = (const float*)d_in[16];
    const float* whh1   = (const float*)d_in[17];
    const float* bih1   = (const float*)d_in[18];
    const float* bhh1   = (const float*)d_in[19];

    float* pred    = (float*)d_out;                          // [B, VOCAB]
    float* hid_out = pred + (long long)B_SZ * VOCAB;         // [2, B, HID]
    float* h0_out  = hid_out;
    float* h1_out  = hid_out + (long long)B_SZ * HID;

    float* ws     = (float*)d_ws;
    float* x_cat  = ws;                                       // B*2H
    float* x_comb = x_cat  + (long long)B_SZ * 2 * HID;       // B*H
    float* gx     = x_comb + (long long)B_SZ * HID;           // B*3H
    float* gh     = gx     + (long long)B_SZ * 3 * HID;       // B*3H
    float* hlout  = gh     + (long long)B_SZ * 3 * HID;       // B*H

    const dim3 blk(256);
    const int  mtiles = B_SZ / 256;                           // 8
    const int  gate_blocks = (int)(((long long)B_SZ * HID + 255) / 256);

    // 1) embedding + attention + concat
    attn_front<<<dim3(B_SZ), blk, 0, stream>>>(rel, hidden, enc, emb,
                                               attn_w, attn_b, x_cat);

    // 2) x = relu(x_cat @ comb_w.T + comb_b)
    gemm_wmma_bf16<<<dim3(HID / 64, mtiles), blk, 0, stream>>>(
        x_cat, 2 * HID, comb_w, HID, comb_b, x_comb, 1);

    // 3) GRU layer 0
    gemm_wmma_bf16<<<dim3(3 * HID / 64, mtiles), blk, 0, stream>>>(
        x_comb, HID, wih0, 3 * HID, bih0, gx, 0);
    gemm_wmma_bf16<<<dim3(3 * HID / 64, mtiles), blk, 0, stream>>>(
        hidden, HID, whh0, 3 * HID, bhh0, gh, 0);
    gru_gate<<<dim3(gate_blocks), blk, 0, stream>>>(gx, gh, hidden, h0_out);

    // 4) GRU layer 1
    gemm_wmma_bf16<<<dim3(3 * HID / 64, mtiles), blk, 0, stream>>>(
        h0_out, HID, wih1, 3 * HID, bih1, gx, 0);
    gemm_wmma_bf16<<<dim3(3 * HID / 64, mtiles), blk, 0, stream>>>(
        hidden + (long long)B_SZ * HID, HID, whh1, 3 * HID, bhh1, gh, 0);
    gru_gate<<<dim3(gate_blocks), blk, 0, stream>>>(
        gx, gh, hidden + (long long)B_SZ * HID, h1_out);

    // 5) out = h1 @ hl_w.T + hl_b
    gemm_wmma_bf16<<<dim3(HID / 64, mtiles), blk, 0, stream>>>(
        h1_out, HID, hl_w, HID, hl_b, hlout, 0);

    // 6) prediction = out @ fc_w.T + fc_b   (N = 20000, guarded tiles)
    gemm_wmma_bf16<<<dim3((VOCAB + 63) / 64, mtiles), blk, 0, stream>>>(
        hlout, HID, fc_w, VOCAB, fc_b, pred, 0);
}
